// SparseMLP_82334523065003
// MI455X (gfx1250) — matrix operand order repaired and checked by reading, not compiled, not run on hardware
//
#include <hip/hip_runtime.h>

// ---------------- problem constants (from reference) ----------------
#define S_   2048
#define B_   4
#define D_   1024
#define E_   8
#define C_   2048
#define FF_  4096
#define N_   (S_ * B_)   // 8192 tokens

// ---------------- vector types for WMMA ----------------
typedef __attribute__((ext_vector_type(16))) __bf16 v16bf;
typedef __attribute__((ext_vector_type(8)))  __bf16 v8bf;
typedef __attribute__((ext_vector_type(4)))  __bf16 v4bf;
typedef __attribute__((ext_vector_type(8)))  float  v8f;

union ABfrag { v16bf v; v8bf h[2]; };

// LDS B-tile geometry: 64 N-rows x 64 K-halves per chunk, row padded to 72
// halves (144B: 16B-aligned, and r*144/4 mod 64 hits 16 distinct bank offsets
// for the 16 fragment-reading lanes). Double buffered: 2*64*72*2B = 18 KB.
#define LDS_ROW  72
#define LDS_BUF  (64 * LDS_ROW)

__device__ __forceinline__ float gelu_tanh(float x) {
    // jax.nn.gelu default (approximate=True)
    const float k0 = 0.7978845608028654f;
    const float k1 = 0.044715f;
    return 0.5f * x * (1.0f + tanhf(k0 * (x + k1 * x * x * x)));
}

// ---------------- kernel 0: zero the output (scatter-add target) ----------------
__global__ void zero_f4(float4* __restrict__ p, long long n4) {
    long long i = (long long)blockIdx.x * blockDim.x + threadIdx.x;
    if (i < n4) p[i] = make_float4(0.f, 0.f, 0.f, 0.f);
}

// ---------------- kernel 1: convert f32 -> bf16 with per-expert transpose ----------------
// src: [E][rows][cols] f32   dst: [E][cols][rows] bf16
__global__ void cvt_transpose(const float* __restrict__ src, __bf16* __restrict__ dst,
                              int rows, int cols) {
    long long idx = (long long)blockIdx.x * blockDim.x + threadIdx.x;
    long long per = (long long)rows * cols;
    long long tot = (long long)E_ * per;
    if (idx >= tot) return;
    int       e   = (int)(idx / per);
    long long rem = idx - (long long)e * per;
    int rr = (int)(rem / cols);
    int cc = (int)(rem - (long long)rr * cols);
    dst[(long long)e * per + (long long)cc * rows + rr] = (__bf16)src[idx];
}

// ---------------- kernel 2: dispatch gather + cast: Abf[e][c][d] = bf16(x[tok[e][c]][d]) ---
__global__ void gather_cast(const float* __restrict__ x,
                            const long long* __restrict__ top_idx, // [C][E]
                            __bf16* __restrict__ Abf) {
    long long idx = (long long)blockIdx.x * blockDim.x + threadIdx.x; // over E*C*(D/4)
    const long long tot = (long long)E_ * C_ * (D_ / 4);
    if (idx >= tot) return;
    int d4 = (int)(idx % (D_ / 4));
    long long rem = idx / (D_ / 4);
    int c = (int)(rem % C_);
    int e = (int)(rem / C_);
    long long tok = top_idx[(long long)c * E_ + e];
    const float4 v = *(const float4*)(x + tok * D_ + 4 * (long long)d4);
    v4bf o; o.x = (__bf16)v.x; o.y = (__bf16)v.y; o.z = (__bf16)v.z; o.w = (__bf16)v.w;
    *(v4bf*)(Abf + ((long long)e * C_ + c) * D_ + 4 * (long long)d4) = o;
}

// ---------------- WMMA A-fragment loader (ISA 16-bit 16x32 layout, wave32) ----------
// lane r=lane&15 -> row M=r; half=lane>>4; v0..3 hold K=8*half+0..7,
// v4..7 hold K=16+8*half+0..7  -> two contiguous 16B loads from a row-major row.
__device__ __forceinline__ v16bf load_a_frag(const __bf16* __restrict__ row, int kk, int half) {
    ABfrag a;
    a.h[0] = *(const v8bf*)(row + kk + 8 * half);
    a.h[1] = *(const v8bf*)(row + kk + 8 * half + 16);
    return a.v;
}

// ---------------- core: 256x64 block tile, 8 waves x (32M x 64N) each --------------
// A: row-major [*, LDA] bf16 (per-wave private, streamed from global)
// B: N-major  [*, LDB] bf16 (shared: staged global -> LDS, double buffered)
template<int KTOT, int LDA, int LDB>
__device__ __forceinline__ void wmma_tile_256x64(
    const __bf16* __restrict__ Ae, const __bf16* __restrict__ Be,
    int mBlock, int n0, __bf16* __restrict__ lds, v8f (&acc)[2][4]) {

    const int tid  = threadIdx.x;
    const int wave = tid >> 5;
    const int lane = tid & 31;
    const int r    = lane & 15;
    const int hf   = lane >> 4;
    const int mW   = mBlock + wave * 32;

    const __bf16* __restrict__ arow0 = Ae + (long long)(mW + r) * LDA;
    const __bf16* __restrict__ arow1 = arow0 + (long long)16 * LDA;

    // cooperative staging coords: thread -> (n row, 16-half quarter) of the chunk
    const int cn = tid >> 2;      // 0..63
    const int cq = tid & 3;       // 0..3
    const __bf16* __restrict__ srcB = Be + (long long)(n0 + cn) * LDB + 16 * cq;
    __bf16* __restrict__ dstB = lds + cn * LDS_ROW + 16 * cq;

    // prologue: stage chunk 0 into buffer 0
    {
        v8bf s0 = *(const v8bf*)(srcB);
        v8bf s1 = *(const v8bf*)(srcB + 8);
        *(v8bf*)(dstB)     = s0;
        *(v8bf*)(dstB + 8) = s1;
    }

    constexpr int NCHUNK = KTOT / 64;
    #pragma unroll 1
    for (int kc = 0; kc < NCHUNK; ++kc) {
        const int cur = kc & 1;
        __syncthreads();   // staging of buf[cur] complete; buf[1-cur] readers done

        // prefetch next B chunk to registers while computing
        v8bf s0, s1;
        const bool hasNext = (kc + 1 < NCHUNK);
        if (hasNext) {
            const __bf16* nsrc = srcB + (long long)(kc + 1) * 64;
            s0 = *(const v8bf*)(nsrc);
            s1 = *(const v8bf*)(nsrc + 8);
        }

        const __bf16* __restrict__ bbuf = lds + cur * LDS_BUF;
        #pragma unroll
        for (int ki = 0; ki < 2; ++ki) {
            const int kk = kc * 64 + ki * 32;
            // issue A global loads first (longest latency)
            v16bf a0 = load_a_frag(arow0, kk, hf);
            v16bf a1 = load_a_frag(arow1, kk, hf);
            // load ALL FOUR B fragments into distinct registers so the backend
            // can clause the 8 ds_load_b128 and then burst 8 WMMAs without a
            // per-pair s_wait_dscnt 0x0 drain.
            ABfrag b[4];
            #pragma unroll
            for (int t = 0; t < 4; ++t) {
                const __bf16* bp = bbuf + (16 * t + r) * LDS_ROW + ki * 32 + 8 * hf;
                b[t].h[0] = *(const v8bf*)(bp);
                b[t].h[1] = *(const v8bf*)(bp + 16);
            }
            #pragma unroll
            for (int t = 0; t < 4; ++t) {
                acc[0][t] = __builtin_amdgcn_wmma_f32_16x16x32_bf16(
                    false, a0, false, b[t].v, (short)0, acc[0][t], false, false);
                acc[1][t] = __builtin_amdgcn_wmma_f32_16x16x32_bf16(
                    false, a1, false, b[t].v, (short)0, acc[1][t], false, false);
            }
        }

        if (hasNext) {  // write prefetched chunk to the other buffer
            __bf16* ndst = dstB + (1 - cur) * LDS_BUF;
            *(v8bf*)(ndst)     = s0;
            *(v8bf*)(ndst + 8) = s1;
        }
    }
}

// ---------------- kernel 3: GEMM1  h = gelu(A @ W1t^T + b1), bf16 out ----------------
// A: Abf[e] [C][D], B: W1t[e] [FF][D] (N-major), out: Hbf[e] [C][FF]
__global__ void __launch_bounds__(256)
gemm1_bias_gelu(const __bf16* __restrict__ Abf, const __bf16* __restrict__ W1t,
                const float* __restrict__ b1, __bf16* __restrict__ Hbf) {
    __shared__ __bf16 ldsB[2 * LDS_BUF];
    const int e  = blockIdx.z;
    const int n0 = blockIdx.x * 64;
    const int mB = blockIdx.y * 256;

    const __bf16* __restrict__ Ae = Abf + (long long)e * C_ * (long long)D_;
    const __bf16* __restrict__ Be = W1t + (long long)e * FF_ * (long long)D_;

    v8f acc[2][4];
    #pragma unroll
    for (int f = 0; f < 2; ++f)
        #pragma unroll
        for (int t = 0; t < 4; ++t) acc[f][t] = {};

    wmma_tile_256x64<D_, D_, D_>(Ae, Be, mB, n0, ldsB, acc);

    const int wave = threadIdx.x >> 5, lane = threadIdx.x & 31;
    const int r = lane & 15, hf = lane >> 4;
    __bf16* __restrict__ He = Hbf + (long long)e * C_ * (long long)FF_;
    #pragma unroll
    for (int t = 0; t < 4; ++t) {
        const int   n    = n0 + 16 * t + r;
        const float bias = b1[(long long)e * FF_ + n];
        #pragma unroll
        for (int f = 0; f < 2; ++f) {
            #pragma unroll
            for (int j = 0; j < 8; ++j) {
                const int m = mB + wave * 32 + 16 * f + 8 * hf + j; // C/D layout
                float v = acc[f][t][j] + bias;
                He[(long long)m * FF_ + n] = (__bf16)gelu_tanh(v);
            }
        }
    }
}

// ---------------- kernel 4: GEMM2 + weighted atomic scatter-add combine -------------
// A: Hbf[e] [C][FF], B: W2t[e] [D][FF] (N-major), out += w * (A@B^T + b2)
__global__ void __launch_bounds__(256)
gemm2_combine(const __bf16* __restrict__ Hbf, const __bf16* __restrict__ W2t,
              const float* __restrict__ b2,
              const long long* __restrict__ top_idx,      // [C][E]
              const float* __restrict__ cw,               // [N][E]
              float* __restrict__ out) {
    __shared__ __bf16 ldsB[2 * LDS_BUF];
    const int e  = blockIdx.z;
    const int n0 = blockIdx.x * 64;                        // over D
    const int mB = blockIdx.y * 256;                       // over C

    const __bf16* __restrict__ Ae = Hbf + (long long)e * C_ * (long long)FF_;
    const __bf16* __restrict__ Be = W2t + (long long)e * D_ * (long long)FF_;

    v8f acc[2][4];
    #pragma unroll
    for (int f = 0; f < 2; ++f)
        #pragma unroll
        for (int t = 0; t < 4; ++t) acc[f][t] = {};

    wmma_tile_256x64<FF_, FF_, FF_>(Ae, Be, mB, n0, ldsB, acc);

    const int wave = threadIdx.x >> 5, lane = threadIdx.x & 31;
    const int r = lane & 15, hf = lane >> 4;

    // per-lane token ids / gate weights for the 16 M rows this lane owns
    long long tok[2][8];
    float     wgt[2][8];
    #pragma unroll
    for (int f = 0; f < 2; ++f) {
        #pragma unroll
        for (int j = 0; j < 8; ++j) {
            const int m = mB + wave * 32 + 16 * f + 8 * hf + j;
            long long tk = top_idx[(long long)m * E_ + e];
            tok[f][j] = tk;
            wgt[f][j] = cw[tk * E_ + e];
        }
    }

    #pragma unroll
    for (int t = 0; t < 4; ++t) {
        const int   n    = n0 + 16 * t + r;
        const float bias = b2[(long long)e * D_ + n];
        #pragma unroll
        for (int f = 0; f < 2; ++f) {
            #pragma unroll
            for (int j = 0; j < 8; ++j) {
                float v = (acc[f][t][j] + bias) * wgt[f][j];
                unsafeAtomicAdd(out + tok[f][j] * D_ + n, v); // global_atomic_add_f32
            }
        }
    }
}

// ---------------- launcher ----------------
extern "C" void kernel_launch(void* const* d_in, const int* in_sizes, int n_in,
                              void* d_out, int out_size, void* d_ws, size_t ws_size,
                              hipStream_t stream) {
    const float*     hidden = (const float*)d_in[0];       // [S][B][D] == x [N][D]
    const float*     cw     = (const float*)d_in[1];       // [N][E]
    const long long* top    = (const long long*)d_in[2];   // [C][E] int64
    const float*     W1     = (const float*)d_in[3];       // [E][D][FF]
    const float*     b1     = (const float*)d_in[4];       // [E][FF]
    const float*     W2     = (const float*)d_in[5];       // [E][FF][D]
    const float*     b2     = (const float*)d_in[6];       // [E][D]
    float*           out    = (float*)d_out;               // [S][B][D]

    // ---- workspace carve-out (256B aligned regions) ----
    char*  ws  = (char*)d_ws;
    size_t off = 0;
    auto carve = [&](size_t bytes) {
        void* p = ws + off;
        off = (off + bytes + 255) & ~(size_t)255;
        return p;
    };
    __bf16* W1t = (__bf16*)carve((size_t)E_ * FF_ * D_ * 2);   // [E][FF][D]
    __bf16* W2t = (__bf16*)carve((size_t)E_ * D_ * FF_ * 2);   // [E][D][FF]
    __bf16* Abf = (__bf16*)carve((size_t)E_ * C_ * D_ * 2);    // [E][C][D]
    __bf16* Hbf = (__bf16*)carve((size_t)E_ * C_ * FF_ * 2);   // [E][C][FF]
    (void)ws_size; (void)in_sizes; (void)n_in; (void)out_size;

    // 0) zero the scatter-add target
    {
        long long n4 = (long long)N_ * D_ / 4;
        zero_f4<<<dim3((unsigned)((n4 + 255) / 256)), dim3(256), 0, stream>>>(
            (float4*)out, n4);
    }
    // 1) weights -> bf16, transposed to N-major for contiguous B loads
    {
        long long tot = (long long)E_ * D_ * FF_;
        unsigned blocks = (unsigned)((tot + 255) / 256);
        cvt_transpose<<<dim3(blocks), dim3(256), 0, stream>>>(W1, W1t, D_, FF_);
        cvt_transpose<<<dim3(blocks), dim3(256), 0, stream>>>(W2, W2t, FF_, D_);
    }
    // 2) MoE dispatch gather + cast
    {
        long long tot = (long long)E_ * C_ * (D_ / 4);
        gather_cast<<<dim3((unsigned)((tot + 255) / 256)), dim3(256), 0, stream>>>(
            hidden, top, Abf);
    }
    // 3) GEMM1 + bias + gelu   (grid: FF/64 x C/256 x E)
    gemm1_bias_gelu<<<dim3(FF_ / 64, C_ / 256, E_), dim3(256), 0, stream>>>(
        Abf, W1t, b1, Hbf);
    // 4) GEMM2 + bias + gate-weighted atomic scatter-add   (grid: D/64 x C/256 x E)
    gemm2_combine<<<dim3(D_ / 64, C_ / 256, E_), dim3(256), 0, stream>>>(
        Hbf, W2t, b2, top, cw, out);
}
